// GCN_24876450578551
// MI455X (gfx1250) — matrix operand order repaired and checked by reading, not compiled, not run on hardware
//
#include <hip/hip_runtime.h>
#include <stdint.h>

// ---------------------------------------------------------------------------
// GCN 3-layer forward for MI455X (gfx1250, wave32).
//  - GEMMs via V_WMMA_F32_16X16X32_F16 with 2-term f16 split (~fp32 accuracy)
//  - Edge aggregation via L2-resident global_atomic_add_f32 (feature matrix
//    25.6MB fits the 192MB L2; edges streamed once per layer from HBM)
// ---------------------------------------------------------------------------

typedef __attribute__((ext_vector_type(16))) _Float16 v16h;
typedef __attribute__((ext_vector_type(8)))  float    v8f;

#define WMMA_F16(a, b, c) \
  __builtin_amdgcn_wmma_f32_16x16x32_f16(false, (a), false, (b), (short)0, (c), false, false)

// -------------------------------- utility ----------------------------------

__global__ void zero_f32(float* __restrict__ p, int n) {
  int i = blockIdx.x * blockDim.x + threadIdx.x;
  if (i < n) p[i] = 0.0f;
}

__global__ void deg_count(const long long* __restrict__ ei, float* __restrict__ deg, int E) {
  int e = blockIdx.x * blockDim.x + threadIdx.x;
  if (e < E) {
    long long d = ei[(long long)E + e];
    unsafeAtomicAdd(&deg[d], 1.0f);   // native global_atomic_add_f32
  }
}

__global__ void compute_dis(float* __restrict__ deg, int N) {
  int i = blockIdx.x * blockDim.x + threadIdx.x;
  if (i < N) deg[i] = rsqrtf(deg[i] + 1.0f);   // +1 self loop; always > 0
}

__global__ void relu_ip(float* __restrict__ x, long long n) {
  long long i = (long long)blockIdx.x * blockDim.x + threadIdx.x;
  if (i < n) x[i] = fmaxf(x[i], 0.0f);
}

// ---------------------- WMMA GEMM: out[N,Nout] = X[N,K] @ W[K,Nout] --------
// Block: 128 threads = 4 waves; wave w computes rows [blk*64 + w*16, +16).
// W is packed into LDS as per-lane v16h fragments (hi/lo f16 split), following
// the CDNA5 16-bit B 32x16 layout: lane L holds col N=L%16, K = (L/16)*8+0..7, 16+(L/16)*8+0..7.
// A fragments per the 16-bit A 16x32 layout: lane L row M=L%16,
//   v0..v3: K = (L/16)*8 + 0..7,  v4..v7: K = 16 + (L/16)*8 + 0..7.

template <int NUMJ, int KSTAGE>
__global__ __launch_bounds__(128)
void gemm_wmma(const float* __restrict__ X, const float* __restrict__ W,
               float* __restrict__ out, int N, int K, int Nout) {
  extern __shared__ __align__(32) _Float16 smem[];
  _Float16* fhi = smem;                                // NUMJ*KSTAGE*32*16 halves
  _Float16* flo = smem + NUMJ * KSTAGE * 32 * 16;

  const int tid  = threadIdx.x;
  const int lane = tid & 31;
  const int wv   = tid >> 5;
  const int half = lane >> 4;   // 0: lanes 0-15, 1: lanes 16-31
  const int mrow = lane & 15;

  const int rowBase = blockIdx.x * 64 + wv * 16;
  int xrow = rowBase + mrow;
  if (xrow > N - 1) xrow = N - 1;                      // clamp (stores masked)
  const float* __restrict__ xr = X + (size_t)xrow * K;

  v8f acc[NUMJ] = {};

  for (int k0 = 0; k0 < K; k0 += KSTAGE * 32) {
    // ---- cooperatively pack W fragments (f16 hi/lo) into LDS ----
    for (int t = tid; t < NUMJ * KSTAGE * 32; t += 128) {
      int L  = t & 31;
      int fs = t >> 5;
      int s  = fs % KSTAGE;
      int j  = fs / KSTAGE;
      int col = j * 16 + (L & 15);
      int kb  = k0 + s * 32 + (L >> 4) * 8;
      _Float16* hp = fhi + (size_t)t * 16;
      _Float16* lp = flo + (size_t)t * 16;
#pragma unroll
      for (int k = 0; k < 16; k++) {
        float w = (col < Nout) ? W[(size_t)(kb + k + (k >= 8 ? 8 : 0)) * Nout + col] : 0.0f;
        _Float16 h = (_Float16)w;
        hp[k] = h;
        lp[k] = (_Float16)(w - (float)h);
      }
    }
    __syncthreads();

    // ---- compute ----
#pragma unroll
    for (int s = 0; s < KSTAGE; s++) {
      const int ks = k0 + s * 32;
      v16h ahi, alo;
#pragma unroll
      for (int i = 0; i < 4; i++) {                    // v0..v3
        float2 p = *(const float2*)(xr + ks + half * 8 + 2 * i);
        _Float16 h0 = (_Float16)p.x, h1 = (_Float16)p.y;
        ahi[2 * i] = h0;                 ahi[2 * i + 1] = h1;
        alo[2 * i] = (_Float16)(p.x - (float)h0);
        alo[2 * i + 1] = (_Float16)(p.y - (float)h1);
      }
#pragma unroll
      for (int i = 0; i < 4; i++) {                    // v4..v7
        float2 p = *(const float2*)(xr + ks + 16 + half * 8 + 2 * i);
        _Float16 h0 = (_Float16)p.x, h1 = (_Float16)p.y;
        ahi[8 + 2 * i] = h0;             ahi[8 + 2 * i + 1] = h1;
        alo[8 + 2 * i] = (_Float16)(p.x - (float)h0);
        alo[8 + 2 * i + 1] = (_Float16)(p.y - (float)h1);
      }
#pragma unroll
      for (int j = 0; j < NUMJ; j++) {
        const v16h bh = *(const v16h*)(fhi + (size_t)((j * KSTAGE + s) * 32 + lane) * 16);
        const v16h bl = *(const v16h*)(flo + (size_t)((j * KSTAGE + s) * 32 + lane) * 16);
        acc[j] = WMMA_F16(ahi, bh, acc[j]);   // hi*hi
        acc[j] = WMMA_F16(ahi, bl, acc[j]);   // hi*lo
        acc[j] = WMMA_F16(alo, bh, acc[j]);   // lo*hi
      }
    }
    __syncthreads();
  }

  // ---- store D: lane L holds col = j*16 + L%16, rows rowBase + (L/16)*8 + r
#pragma unroll
  for (int j = 0; j < NUMJ; j++) {
    int col = j * 16 + mrow;
    if (col < Nout) {
#pragma unroll
      for (int r = 0; r < 8; r++) {
        int row = rowBase + half * 8 + r;
        if (row < N) out[(size_t)row * Nout + col] = acc[j][r];
      }
    }
  }
}

// -------------------- aggregation: self-loop + bias init -------------------
// out[i,f] = h[i,f]*dis[i]^2 + b[f]   (self-loop contribution fused with bias)

template <int F>
__global__ void init_selfloop_bias(const float* __restrict__ h, const float* __restrict__ dis,
                                   const float* __restrict__ b, float* __restrict__ out, int N) {
  long long idx = (long long)blockIdx.x * blockDim.x + threadIdx.x;
  if (idx >= (long long)N * F) return;
  int i = (int)(idx / F);
  int f = (int)(idx % F);
  float s = dis[i];
  out[idx] = h[idx] * s * s + b[f];
}

// -------------------- aggregation: edge scatter-add ------------------------
// One thread per (edge, 4-float chunk).  Gather h[src] (L2-resident),
// scale by dis[src]*dis[dst], scatter-add with native f32 atomics.

template <int F, int CH>
__global__ void edge_scatter(const float* __restrict__ h, const long long* __restrict__ ei,
                             const float* __restrict__ dis, float* __restrict__ out, int E) {
  long long idx = (long long)blockIdx.x * blockDim.x + threadIdx.x;
  if (idx >= (long long)E * CH) return;
  long long e = idx / CH;
  int c = (int)(idx % CH);
  long long s = ei[e];
  long long d = ei[(long long)E + e];
  // Stream-ahead prefetch of the edge list (HBM-resident, read sequentially).
  if ((e & 1023) == 0 && (e + 8192) < (long long)E) {
    __builtin_prefetch(&ei[e + 8192], 0, 1);
    __builtin_prefetch(&ei[(long long)E + e + 8192], 0, 1);
  }
  float nrm = dis[s] * dis[d];
  const float4 v = *(const float4*)(h + s * F + c * 4);
  float* ob = out + d * F + c * 4;
  unsafeAtomicAdd(ob + 0, v.x * nrm);
  unsafeAtomicAdd(ob + 1, v.y * nrm);
  unsafeAtomicAdd(ob + 2, v.z * nrm);
  unsafeAtomicAdd(ob + 3, v.w * nrm);
}

// ---------------------------------------------------------------------------

extern "C" void kernel_launch(void* const* d_in, const int* in_sizes, int n_in,
                              void* d_out, int out_size, void* d_ws, size_t ws_size,
                              hipStream_t stream) {
  const float*     X  = (const float*)d_in[0];
  const long long* ei = (const long long*)d_in[1];   // edge_index [2,E] int64
  const float*     W1 = (const float*)d_in[2];
  const float*     b1 = (const float*)d_in[3];
  const float*     W2 = (const float*)d_in[4];
  const float*     b2 = (const float*)d_in[5];
  const float*     W3 = (const float*)d_in[6];
  const float*     b3 = (const float*)d_in[7];

  const int F = 256, H = 64, C = 40;
  const int N = in_sizes[0] / F;     // 100000
  const int E = in_sizes[1] / 2;     // 3200000

  // workspace: dis[N] | bufA[N*64] | bufB[N*64]  (~52 MB)
  char* ws = (char*)d_ws;
  float* dis = (float*)ws;
  size_t o1 = ((size_t)N * sizeof(float) + 255) & ~(size_t)255;
  float* bufA = (float*)(ws + o1);
  size_t o2 = o1 + (((size_t)N * H * sizeof(float) + 255) & ~(size_t)255);
  float* bufB = (float*)(ws + o2);
  float* outp = (float*)d_out;

  const int T = 256;
  const int gblk = (N + 63) / 64;

  // degree -> dis = rsqrt(deg+1)
  zero_f32<<<(N + T - 1) / T, T, 0, stream>>>(dis, N);
  deg_count<<<(E + T - 1) / T, T, 0, stream>>>(ei, dis, E);
  compute_dis<<<(N + T - 1) / T, T, 0, stream>>>(dis, N);

  // ---- layer 1: h = X @ W1 ; agg ; +b1 ; relu ----
  {
    const size_t sh = (size_t)4 * 4 * 32 * 16 * sizeof(_Float16) * 2;   // 32 KB
    gemm_wmma<4, 4><<<gblk, 128, sh, stream>>>(X, W1, bufA, N, F, H);
  }
  init_selfloop_bias<64><<<(int)(((long long)N * 64 + T - 1) / T), T, 0, stream>>>(bufA, dis, b1, bufB, N);
  edge_scatter<64, 16><<<(int)(((long long)E * 16 + T - 1) / T), T, 0, stream>>>(bufA, ei, dis, bufB, E);
  relu_ip<<<(int)(((long long)N * 64 + T - 1) / T), T, 0, stream>>>(bufB, (long long)N * 64);

  // ---- layer 2: h = agg1 @ W2 ; agg ; +b2 ; relu ----
  {
    const size_t sh = (size_t)4 * 2 * 32 * 16 * sizeof(_Float16) * 2;   // 16 KB
    gemm_wmma<4, 2><<<gblk, 128, sh, stream>>>(bufB, W2, bufA, N, H, H);
  }
  init_selfloop_bias<64><<<(int)(((long long)N * 64 + T - 1) / T), T, 0, stream>>>(bufA, dis, b2, bufB, N);
  edge_scatter<64, 16><<<(int)(((long long)E * 16 + T - 1) / T), T, 0, stream>>>(bufA, ei, dis, bufB, E);
  relu_ip<<<(int)(((long long)N * 64 + T - 1) / T), T, 0, stream>>>(bufB, (long long)N * 64);

  // ---- layer 3: h = agg2 @ W3 ; agg ; +b3 (no relu) ----
  {
    const size_t sh = (size_t)3 * 2 * 32 * 16 * sizeof(_Float16) * 2;   // 12 KB
    gemm_wmma<3, 2><<<gblk, 128, sh, stream>>>(bufB, W3, bufA, N, H, C);
  }
  init_selfloop_bias<40><<<(int)(((long long)N * 40 + T - 1) / T), T, 0, stream>>>(bufA, dis, b3, outp, N);
  edge_scatter<40, 10><<<(int)(((long long)E * 10 + T - 1) / T), T, 0, stream>>>(bufA, ei, dis, outp, E);
}
